// BiLSTM_Linear_Classification_Pretrained_28750511079844
// MI455X (gfx1250) — compile-verified, matrix operand order checked by reading, not run on hardware
//
#include <hip/hip_runtime.h>
#include <hip/hip_bf16.h>

// ---------------------------------------------------------------------------
// MI455X (gfx1250) implementation.
// Precision: f16 WMMA (v_wmma_f32_16x16x32_f16) with f32 accumulation.
// Model (~46MB in f16) is fully L2-resident (192MB L2); the LSTM recurrence
// is the serial bottleneck, so it runs as one workgroup (32 waves, wave32)
// per direction with h in LDS (padded to M=16) and c in registers.
// GEMM blocks stage their shared A strip into LDS with the CDNA5 async
// copy engine (global_load_async_to_lds_b128 / s_wait_asynccnt).
// ---------------------------------------------------------------------------

typedef __attribute__((ext_vector_type(16))) _Float16 v16h;
typedef __attribute__((ext_vector_type(8)))  _Float16 v8h;
typedef __attribute__((ext_vector_type(8)))  float    v8f;

#define WMMA_F16(a, b, c) \
  __builtin_amdgcn_wmma_f32_16x16x32_f16(false, (a), false, (b), (short)0, (c), false, false)

// ---- WMMA fragment loaders (ISA 7.12.2 layouts, wave32) -------------------
// A (16x32, row-major, M x K) and B-from-W^T (W stored N x K row-major) share
// the same per-lane layout: lane%16 -> row, lane/16 selects K-halves {0..7,
// 16..23} vs {8..15, 24..31}.
static __device__ __forceinline__ v16h frag_rm(const _Float16* base, int ld,
                                               int row0, int k0) {
  const int lane = threadIdx.x & 31;
  const _Float16* p =
      base + (size_t)(row0 + (lane & 15)) * ld + (size_t)(k0 + ((lane >> 4) << 3));
  v8h lo = *(const v8h*)p;
  v8h hi = *(const v8h*)(p + 16);
  return __builtin_shufflevector(lo, hi, 0, 1, 2, 3, 4, 5, 6, 7, 8, 9, 10, 11,
                                 12, 13, 14, 15);
}

// B fragment from a NON-transposed K x N row-major matrix (strided gathers).
static __device__ __forceinline__ v16h frag_nt(const _Float16* base, int ld,
                                               int k0, int n0) {
  const int lane = threadIdx.x & 31;
  const int n = n0 + (lane & 15);
  const int kb = k0 + ((lane >> 4) << 3);
  v16h r;
#pragma unroll
  for (int j = 0; j < 8; ++j) {
    r[j] = base[(size_t)(kb + j) * ld + n];
    r[j + 8] = base[(size_t)(kb + 16 + j) * ld + n];
  }
  return r;
}

// ---- GEMM: C(M,N) = act( A(M,K) * W(N,K)^T + bias ) -----------------------
// block = 256 threads = 8 waves; each wave owns one 16x16 C tile; grid.x
// covers N/128, grid.y covers M/16, grid.z = batch (z -> (zb, zh)).
// The A strip (16 x K) is shared by all 8 waves -> staged once into LDS via
// the async copy engine; A fragments then come from LDS (ds_load_b128).
__global__ void k_gemm_tn(const _Float16* __restrict__ A, int lda, long sAb, long sAh,
                          const _Float16* __restrict__ W, int ldw, long sWb, long sWh,
                          const float* __restrict__ bias, long sBiasZ,
                          float* __restrict__ Cf, _Float16* __restrict__ Ch,
                          int ldc, long sCb, long sCh,
                          int M, int N, int K, int nH, int act) {
  extern __shared__ char smemA[];  // 16 * K * 2 bytes, offset 0 of LDS alloc
  const int z = blockIdx.z, zb = z / nH, zh = z % nH;
  const _Float16* Ap = A + (size_t)zb * sAb + (size_t)zh * sAh;
  const _Float16* Wp = W + (size_t)zb * sWb + (size_t)zh * sWh;
  const size_t co = (size_t)zb * sCb + (size_t)zh * sCh;
  const int m0 = blockIdx.y * 16;

  // --- async stage of the A strip: 16 rows x K cols (16B chunks per lane) ---
  {
    const int cpr = K >> 3;  // 16-byte chunks per row
    for (int c = threadIdx.x; c < 16 * cpr; c += blockDim.x) {
      const int row = c / cpr, cc = c - row * cpr;
      const _Float16* g = Ap + (size_t)(m0 + row) * lda + (cc << 3);
      const unsigned lds_off = (unsigned)(((row * K) + (cc << 3)) * 2);
      asm volatile("global_load_async_to_lds_b128 %0, %1, off"
                   :: "v"(lds_off), "v"((unsigned long long)(uintptr_t)g)
                   : "memory");
    }
    asm volatile("s_wait_asynccnt 0x0" ::: "memory");
  }
  __syncthreads();
  const _Float16* As = (const _Float16*)smemA;

  const int n0 = (blockIdx.x * 8 + (threadIdx.x >> 5)) * 16;
  if (n0 < N) {
    v8f acc = {};
    for (int k0 = 0; k0 < K; k0 += 32) {
      v16h a = frag_rm(As, K, 0, k0);       // ds_load from LDS
      v16h b = frag_rm(Wp, ldw, n0, k0);    // L2-resident weight stream
      __builtin_prefetch((const void*)(Wp + (size_t)n0 * ldw + k0 + 256), 0, 0);
      acc = WMMA_F16(a, b, acc);
    }
    const int lane = threadIdx.x & 31;
    const int col = n0 + (lane & 15);
    const int rbase = m0 + ((lane >> 4) << 3);
    const float bv = bias ? bias[(size_t)z * sBiasZ + col] : 0.f;
#pragma unroll
    for (int i = 0; i < 8; ++i) {
      const int row = rbase + i;
      if (row < M) {
        float v = acc[i] + bv;
        if (act == 1) v = fmaxf(v, 0.f);
        if (Cf) Cf[co + (size_t)row * ldc + col] = v;
        if (Ch) Ch[co + (size_t)row * ldc + col] = (_Float16)v;
      }
    }
  }
}

// ---- GEMM: C(M,N) = A(M,K) * B(K,N)   (B non-transposed) ------------------
__global__ void k_gemm_nn(const _Float16* __restrict__ A, int lda, long sAb, long sAh,
                          const _Float16* __restrict__ Bm, int ldb, long sBb, long sBh,
                          float* __restrict__ Cf, _Float16* __restrict__ Ch,
                          int ldc, long sCb, long sCh,
                          int M, int N, int K, int nH) {
  const int z = blockIdx.z, zb = z / nH, zh = z % nH;
  const _Float16* Ap = A + (size_t)zb * sAb + (size_t)zh * sAh;
  const _Float16* Bp = Bm + (size_t)zb * sBb + (size_t)zh * sBh;
  const size_t co = (size_t)zb * sCb + (size_t)zh * sCh;
  const int n0 = (blockIdx.x * 8 + (threadIdx.x >> 5)) * 16;
  const int m0 = blockIdx.y * 16;
  if (n0 >= N) return;
  v8f acc = {};
  for (int k0 = 0; k0 < K; k0 += 32) {
    v16h a = frag_rm(Ap, lda, m0, k0);
    v16h b = frag_nt(Bp, ldb, k0, n0);
    acc = WMMA_F16(a, b, acc);
  }
  const int lane = threadIdx.x & 31;
  const int col = n0 + (lane & 15);
  const int rbase = m0 + ((lane >> 4) << 3);
#pragma unroll
  for (int i = 0; i < 8; ++i) {
    const int row = rbase + i;
    if (row < M) {
      float v = acc[i];
      if (Cf) Cf[co + (size_t)row * ldc + col] = v;
      if (Ch) Ch[co + (size_t)row * ldc + col] = (_Float16)v;
    }
  }
}

// ---- BiLSTM recurrence ----------------------------------------------------
// One workgroup (1024 thr = 32 waves) per direction (grid.x = 2).
// gin = X @ Wih^T + b precomputed (f32, [dir][B*S][4H]).
// LDS: hpad (16 x H f16, rows 4..15 = 0) + gate preactivations G (4 x 4H f32).
// Each step: 32 waves WMMA h @ Whh^T into G, barrier, threads j<H apply the
// i,f,g,o nonlinearities with c[b][j] held in registers, write h back to LDS.
__global__ void k_bilstm(const float* __restrict__ gin, long gin_dir_stride,
                         const _Float16* __restrict__ Whh, long whh_dir_stride,
                         _Float16* __restrict__ out, long out_dir_stride,
                         int out_ld, int col_per_dir, int H, int S) {
  extern __shared__ char smem[];
  const int G4 = 4 * H;
  _Float16* hpad = (_Float16*)smem;                       // 16 x H
  float* G = (float*)(smem + (size_t)16 * H * 2);         // 4 x 4H
  const int dir = blockIdx.x;
  const float* ginD = gin + (size_t)dir * gin_dir_stride;
  const _Float16* Wd = Whh + (size_t)dir * whh_dir_stride;
  _Float16* outD = out + (size_t)dir * out_dir_stride + (size_t)dir * col_per_dir;

  for (int i = threadIdx.x; i < 16 * H; i += blockDim.x) hpad[i] = (_Float16)0.f;
  __syncthreads();

  float c0 = 0.f, c1 = 0.f, c2 = 0.f, c3 = 0.f;
  const int wave = threadIdx.x >> 5;
  const int lane = threadIdx.x & 31;
  const int tiles = G4 / 16 / 32;  // N-tiles per wave

  for (int step = 0; step < S; ++step) {
    const int t = dir ? (S - 1 - step) : step;
    // phase 1: G[0:4][:] = hpad @ Whh^T  (rows 4..15 of hpad are zero)
    for (int ti = 0; ti < tiles; ++ti) {
      const int n0 = (wave * tiles + ti) * 16;
      v8f acc = {};
      for (int k0 = 0; k0 < H; k0 += 32) {
        v16h a = frag_rm(hpad, H, 0, k0);     // ds_load from LDS
        v16h b = frag_rm(Wd, H, n0, k0);      // L2-resident weight stream
        acc = WMMA_F16(a, b, acc);
      }
      if (lane < 16) {
#pragma unroll
        for (int i = 0; i < 4; ++i) G[i * G4 + n0 + lane] = acc[i];
      }
    }
    __syncthreads();
    // phase 2: gate nonlinearities, c in registers, h -> LDS + global
    const int j = threadIdx.x;
    if (j < H) {
      float cc[4] = {c0, c1, c2, c3};
#pragma unroll
      for (int b = 0; b < 4; ++b) {
        const float* gr = ginD + ((size_t)(b * S + t)) * G4;
        const float gi = gr[j] + G[b * G4 + j];
        const float gf = gr[H + j] + G[b * G4 + H + j];
        const float gg = gr[2 * H + j] + G[b * G4 + 2 * H + j];
        const float go = gr[3 * H + j] + G[b * G4 + 3 * H + j];
        const float ii = 1.f / (1.f + __expf(-gi));
        const float ff = 1.f / (1.f + __expf(-gf));
        const float oo = 1.f / (1.f + __expf(-go));
        const float c = ff * cc[b] + ii * tanhf(gg);
        const float h = oo * tanhf(c);
        cc[b] = c;
        hpad[b * H + j] = (_Float16)h;
        outD[((size_t)(b * S + t)) * out_ld + j] = (_Float16)h;
      }
      c0 = cc[0]; c1 = cc[1]; c2 = cc[2]; c3 = cc[3];
    }
    __syncthreads();
  }
}

// ---- elementwise / reduction kernels (wave32 shuffle reductions) ----------
__global__ void k_layernorm(const float* __restrict__ x, const float* __restrict__ w,
                            const float* __restrict__ b, _Float16* __restrict__ y,
                            int rows, int cols) {
  const int row = blockIdx.x * (blockDim.x >> 5) + (threadIdx.x >> 5);
  if (row >= rows) return;
  const int lane = threadIdx.x & 31;
  const float* xr = x + (size_t)row * cols;
  float s = 0.f, ss = 0.f;
  for (int c = lane; c < cols; c += 32) { float v = xr[c]; s += v; ss += v * v; }
  for (int m = 16; m > 0; m >>= 1) { s += __shfl_xor(s, m); ss += __shfl_xor(ss, m); }
  const float mu = s / cols;
  const float rs = rsqrtf(ss / cols - mu * mu + 1e-5f);
  _Float16* yr = y + (size_t)row * cols;
  for (int c = lane; c < cols; c += 32)
    yr[c] = (_Float16)(((xr[c] - mu) * rs) * w[c] + b[c]);
}

__global__ void k_softmax_scale(const float* __restrict__ s, _Float16* __restrict__ p,
                                int rows, int cols, float scale) {
  const int row = blockIdx.x * (blockDim.x >> 5) + (threadIdx.x >> 5);
  if (row >= rows) return;
  const int lane = threadIdx.x & 31;
  const float* sr = s + (size_t)row * cols;
  float m = -3.4e38f;
  for (int c = lane; c < cols; c += 32) m = fmaxf(m, sr[c] * scale);
  for (int k = 16; k > 0; k >>= 1) m = fmaxf(m, __shfl_xor(m, k));
  float sum = 0.f;
  for (int c = lane; c < cols; c += 32) sum += __expf(sr[c] * scale - m);
  for (int k = 16; k > 0; k >>= 1) sum += __shfl_xor(sum, k);
  const float inv = 1.f / sum;
  _Float16* pr = p + (size_t)row * cols;
  for (int c = lane; c < cols; c += 32)
    pr[c] = (_Float16)(__expf(sr[c] * scale - m) * inv);
}

__global__ void k_f32_to_f16(const float* __restrict__ a, _Float16* __restrict__ o, long n) {
  const long i = (long)blockIdx.x * blockDim.x + threadIdx.x;
  if (i < n) o[i] = (_Float16)a[i];
}

__global__ void k_add_f16(const _Float16* __restrict__ a, const _Float16* __restrict__ b,
                          _Float16* __restrict__ o, long n) {
  const long i = (long)blockIdx.x * blockDim.x + threadIdx.x;
  if (i < n) o[i] = (_Float16)((float)a[i] + (float)b[i]);
}

__global__ void k_valid_idx(const int* __restrict__ mask, int* __restrict__ idx, int n) {
  if (threadIdx.x == 0 && blockIdx.x == 0) {
    int c = 0;
    for (int i = 0; i < n; ++i)
      if (mask[i] == 1) idx[c++] = i;
  }
}

__global__ void k_gather_rows(const float* __restrict__ a, const int* __restrict__ idx,
                              float* __restrict__ o, int rows, int cols) {
  const long i = (long)blockIdx.x * blockDim.x + threadIdx.x;
  if (i >= (long)rows * cols) return;
  const int r = (int)(i / cols), c = (int)(i % cols);
  o[i] = a[(size_t)idx[r] * cols + c];
}

__global__ void k_bn_stats(const float* __restrict__ e, float* __restrict__ mean,
                           float* __restrict__ rstd, int rows, int cols) {
  const int j = blockIdx.x * blockDim.x + threadIdx.x;
  if (j >= cols) return;
  float s = 0.f, ss = 0.f;
  for (int r = 0; r < rows; ++r) { float v = e[(size_t)r * cols + j]; s += v; ss += v * v; }
  const float mu = s / rows;
  mean[j] = mu;
  rstd[j] = rsqrtf(ss / rows - mu * mu + 1e-5f);
}

__global__ void k_bn_apply_pad(const float* __restrict__ e, const float* __restrict__ mean,
                               const float* __restrict__ rstd, const float* __restrict__ w,
                               const float* __restrict__ b, _Float16* __restrict__ o,
                               int rows, int rowsPad, int cols) {
  const long i = (long)blockIdx.x * blockDim.x + threadIdx.x;
  if (i >= (long)rowsPad * cols) return;
  const int r = (int)(i / cols), c = (int)(i % cols);
  float v = 0.f;
  if (r < rows) v = (e[(size_t)r * cols + c] - mean[c]) * rstd[c] * w[c] + b[c];
  o[i] = (_Float16)v;
}

__global__ void k_mlp2_sigmoid(const float* __restrict__ h, const float* __restrict__ W2,
                               const float* __restrict__ b2, float* __restrict__ out,
                               int rows, int K) {
  const int row = blockIdx.x * (blockDim.x >> 5) + (threadIdx.x >> 5);
  if (row >= rows) return;
  const int lane = threadIdx.x & 31;
  float s = 0.f;
  for (int k = lane; k < K; k += 32) s += h[(size_t)row * K + k] * W2[k];
  for (int m = 16; m > 0; m >>= 1) s += __shfl_xor(s, m);
  if (lane == 0) out[row] = 1.f / (1.f + __expf(-(s + b2[0])));
}

// ---------------------------------------------------------------------------
extern "C" void kernel_launch(void* const* d_in, const int* in_sizes, int n_in,
                              void* d_out, int out_size, void* d_ws, size_t ws_size,
                              hipStream_t stream) {
  constexpr int Bn = 4, S = 256, DIN = 512, H1 = 256, H2 = 512, H3 = 1024, DM = 1024;
  constexpr int BS = Bn * S;  // 1024 rows
  const int nValid = out_size;          // 454
  const int nValidPad = ((nValid + 15) / 16) * 16;  // 464

  const float* x     = (const float*)d_in[0];
  const int*   mask  = (const int*)d_in[1];
  const float* ln_w  = (const float*)d_in[3];
  const float* ln_b  = (const float*)d_in[4];
  const float* Wih1  = (const float*)d_in[5];
  const float* Whh1  = (const float*)d_in[6];
  const float* b1l   = (const float*)d_in[7];
  const float* Wih2  = (const float*)d_in[8];
  const float* Whh2  = (const float*)d_in[9];
  const float* b2l   = (const float*)d_in[10];
  const float* Wih3  = (const float*)d_in[11];
  const float* Whh3  = (const float*)d_in[12];
  const float* b3l   = (const float*)d_in[13];
  const float* Wo1   = (const float*)d_in[14];
  const float* bo1   = (const float*)d_in[15];
  const float* Wo2   = (const float*)d_in[16];
  const float* bo2   = (const float*)d_in[17];
  const float* bn_w  = (const float*)d_in[18];
  const float* bn_b  = (const float*)d_in[19];
  const float* W1    = (const float*)d_in[20];
  const float* b1m   = (const float*)d_in[21];
  const float* W2    = (const float*)d_in[22];
  const float* b2m   = (const float*)d_in[23];
  float* out = (float*)d_out;

  // bump allocator over d_ws
  char* wsb = (char*)d_ws;
  size_t off = 0;
  auto alloc = [&](size_t bytes) -> void* {
    void* p = wsb + off;
    off = (off + bytes + 255) & ~(size_t)255;
    return p;
  };

  _Float16* xln   = (_Float16*)alloc((size_t)BS * DIN * 2);
  _Float16* Wih1h = (_Float16*)alloc((size_t)in_sizes[5] * 2);
  _Float16* Whh1h = (_Float16*)alloc((size_t)in_sizes[6] * 2);
  _Float16* Wih2h = (_Float16*)alloc((size_t)in_sizes[8] * 2);
  _Float16* Whh2h = (_Float16*)alloc((size_t)in_sizes[9] * 2);
  _Float16* Wih3h = (_Float16*)alloc((size_t)in_sizes[11] * 2);
  _Float16* Whh3h = (_Float16*)alloc((size_t)in_sizes[12] * 2);
  _Float16* Wo1h  = (_Float16*)alloc((size_t)in_sizes[14] * 2);
  _Float16* Wo2h  = (_Float16*)alloc((size_t)in_sizes[16] * 2);
  _Float16* W1h   = (_Float16*)alloc((size_t)in_sizes[20] * 2);
  float*    gin1  = (float*)alloc((size_t)2 * BS * 4 * H1 * 4);
  _Float16* l1out = (_Float16*)alloc((size_t)BS * 2 * H1 * 2);
  float*    gin2  = (float*)alloc((size_t)2 * BS * 4 * H2 * 4);
  _Float16* l2dir = (_Float16*)alloc((size_t)2 * BS * H2 * 2);
  _Float16* l2sum = (_Float16*)alloc((size_t)BS * H2 * 2);
  float*    gin3  = (float*)alloc((size_t)2 * BS * 4 * H3 * 4);
  _Float16* l3dir = (_Float16*)alloc((size_t)2 * BS * H3 * 2);
  _Float16* l3sum = (_Float16*)alloc((size_t)BS * H3 * 2);
  float*    scor  = (float*)alloc((size_t)32 * S * S * 4);
  _Float16* Pm    = (_Float16*)alloc((size_t)32 * S * S * 2);
  _Float16* attb  = (_Float16*)alloc((size_t)BS * DM * 2);
  _Float16* a1    = (_Float16*)alloc((size_t)BS * DM * 2);
  float*    a2f   = (float*)alloc((size_t)BS * DM * 4);
  int*      idx   = (int*)alloc((size_t)BS * 4);
  float*    embf  = (float*)alloc((size_t)nValidPad * DM * 4);
  float*    bnMu  = (float*)alloc((size_t)DM * 4);
  float*    bnRs  = (float*)alloc((size_t)DM * 4);
  _Float16* emb16 = (_Float16*)alloc((size_t)nValidPad * DM * 2);
  float*    h1f   = (float*)alloc((size_t)nValidPad * 512 * 4);
  (void)ws_size; (void)n_in;

  auto conv = [&](const float* src, _Float16* dst, long n) {
    k_f32_to_f16<<<dim3((unsigned)((n + 255) / 256)), dim3(256), 0, stream>>>(src, dst, n);
  };

  // 0) LayerNorm -> f16
  k_layernorm<<<dim3(BS / 8), dim3(256), 0, stream>>>(x, ln_w, ln_b, xln, BS, DIN);

  // 1) weights -> f16 (L2-resident thereafter)
  conv(Wih1, Wih1h, in_sizes[5]);  conv(Whh1, Whh1h, in_sizes[6]);
  conv(Wih2, Wih2h, in_sizes[8]);  conv(Whh2, Whh2h, in_sizes[9]);
  conv(Wih3, Wih3h, in_sizes[11]); conv(Whh3, Whh3h, in_sizes[12]);
  conv(Wo1, Wo1h, in_sizes[14]);   conv(Wo2, Wo2h, in_sizes[16]);
  conv(W1, W1h, in_sizes[20]);

  // ---- layer 1 (Din=512 -> H=256, bi) ----
  k_gemm_tn<<<dim3(4 * H1 / 128, BS / 16, 2), dim3(256), 16 * DIN * 2, stream>>>(
      xln, DIN, 0, 0, Wih1h, DIN, (long)4 * H1 * DIN, 0, b1l, 4 * H1,
      gin1, nullptr, 4 * H1, (long)BS * 4 * H1, 0, BS, 4 * H1, DIN, 1, 0);
  k_bilstm<<<dim3(2), dim3(1024), 96 * H1, stream>>>(
      gin1, (long)BS * 4 * H1, Whh1h, (long)4 * H1 * H1,
      l1out, 0, 2 * H1, H1, H1, S);

  // ---- layer 2 (512 -> H=512, bi, dir-sum) ----
  k_gemm_tn<<<dim3(4 * H2 / 128, BS / 16, 2), dim3(256), 16 * 2 * H1 * 2, stream>>>(
      l1out, 2 * H1, 0, 0, Wih2h, 2 * H1, (long)4 * H2 * 2 * H1, 0, b2l, 4 * H2,
      gin2, nullptr, 4 * H2, (long)BS * 4 * H2, 0, BS, 4 * H2, 2 * H1, 1, 0);
  k_bilstm<<<dim3(2), dim3(1024), 96 * H2, stream>>>(
      gin2, (long)BS * 4 * H2, Whh2h, (long)4 * H2 * H2,
      l2dir, (long)BS * H2, H2, 0, H2, S);
  k_add_f16<<<dim3((BS * H2 + 255) / 256), dim3(256), 0, stream>>>(
      l2dir, l2dir + (size_t)BS * H2, l2sum, (long)BS * H2);

  // ---- layer 3 (512 -> H=1024, bi, dir-sum) ----
  k_gemm_tn<<<dim3(4 * H3 / 128, BS / 16, 2), dim3(256), 16 * H2 * 2, stream>>>(
      l2sum, H2, 0, 0, Wih3h, H2, (long)4 * H3 * H2, 0, b3l, 4 * H3,
      gin3, nullptr, 4 * H3, (long)BS * 4 * H3, 0, BS, 4 * H3, H2, 1, 0);
  k_bilstm<<<dim3(2), dim3(1024), 96 * H3, stream>>>(
      gin3, (long)BS * 4 * H3, Whh3h, (long)4 * H3 * H3,
      l3dir, (long)BS * H3, H3, 0, H3, S);
  k_add_f16<<<dim3((BS * H3 + 255) / 256), dim3(256), 0, stream>>>(
      l3dir, l3dir + (size_t)BS * H3, l3sum, (long)BS * H3);

  // ---- attention blocks: scores = q q^T, softmax, P q, proj + relu ----
  const float scale = 0.088388347648318447f;  // 1/sqrt(128)
  auto attention = [&](const _Float16* xin, const _Float16* WoH, const float* bo,
                       _Float16* outH, float* outF) {
    // scores (per b,head): A(256,128) * A^T  -> 32 batched WMMA GEMMs
    k_gemm_tn<<<dim3(S / 128, S / 16, 32), dim3(256), 16 * 128 * 2, stream>>>(
        xin, DM, (long)S * DM, 128, xin, DM, (long)S * DM, 128, nullptr, 0,
        scor, nullptr, S, (long)8 * S * S, (long)S * S, S, S, 128, 8, 0);
    k_softmax_scale<<<dim3(32 * S / 8), dim3(256), 0, stream>>>(scor, Pm, 32 * S, S, scale);
    // out = P(256,256) * Q(256,128), written back into (B,S,1024) head slices
    k_gemm_nn<<<dim3(1, S / 16, 32), dim3(256), 0, stream>>>(
        Pm, S, (long)8 * S * S, (long)S * S, xin, DM, (long)S * DM, 128,
        nullptr, attb, DM, (long)S * DM, 128, S, 128, S, 8);
    // projection + bias + relu
    k_gemm_tn<<<dim3(DM / 128, BS / 16, 1), dim3(256), 16 * DM * 2, stream>>>(
        attb, DM, 0, 0, WoH, DM, 0, 0, bo, 0,
        outF, outH, DM, 0, 0, BS, DM, DM, 1, 1);
  };
  attention(l3sum, Wo1h, bo1, a1, nullptr);    // a1 = relu(attn1(l3sum))
  attention(a1, Wo2h, bo2, nullptr, a2f);      // a2 = relu(attn2(a1)), f32

  // ---- variable-length extraction + BatchNorm (batch stats) ----
  k_valid_idx<<<dim3(1), dim3(32), 0, stream>>>(mask, idx, BS);
  k_gather_rows<<<dim3((nValid * DM + 255) / 256), dim3(256), 0, stream>>>(
      a2f, idx, embf, nValid, DM);
  k_bn_stats<<<dim3(DM / 256), dim3(256), 0, stream>>>(embf, bnMu, bnRs, nValid, DM);
  k_bn_apply_pad<<<dim3((nValidPad * DM + 255) / 256), dim3(256), 0, stream>>>(
      embf, bnMu, bnRs, bn_w, bn_b, emb16, nValid, nValidPad, DM);

  // ---- MLP head: relu(emb @ W1^T + b1) -> sigmoid(h @ W2^T + b2) ----
  k_gemm_tn<<<dim3(512 / 128, nValidPad / 16, 1), dim3(256), 16 * DM * 2, stream>>>(
      emb16, DM, 0, 0, W1h, DM, 0, 0, b1m, 0,
      h1f, nullptr, 512, 0, 0, nValid, 512, DM, 1, 1);
  k_mlp2_sigmoid<<<dim3((nValid + 7) / 8), dim3(256), 0, stream>>>(
      h1f, W2, b2m, out, nValid, 512);
}
